// DecoderBlock_3040836846189
// MI455X (gfx1250) — matrix operand order repaired and checked by reading, not compiled, not run on hardware
//
#include <hip/hip_runtime.h>
#include <math.h>

#define B_  64
#define T_  256
#define C_  256
#define H_  16
#define HS_ 16
#define FF_ 1024
#define M_  (B_ * T_)  // 16384

typedef __attribute__((ext_vector_type(16))) __bf16 v16bf;
typedef __attribute__((ext_vector_type(8)))  float  v8f;

union AF { v16bf v; uint4 q[2]; };

#define WMMA_BF16(Am, Bm, Cm) \
  __builtin_amdgcn_wmma_f32_16x16x32_bf16(false, (Am), false, (Bm), (short)0, (Cm), false, false)

__device__ __forceinline__ float half_max16(float v) {
  v = fmaxf(v, __shfl_xor(v, 1, 32));
  v = fmaxf(v, __shfl_xor(v, 2, 32));
  v = fmaxf(v, __shfl_xor(v, 4, 32));
  v = fmaxf(v, __shfl_xor(v, 8, 32));
  return v;
}
__device__ __forceinline__ float half_sum16(float v) {
  v += __shfl_xor(v, 1, 32);
  v += __shfl_xor(v, 2, 32);
  v += __shfl_xor(v, 4, 32);
  v += __shfl_xor(v, 8, 32);
  return v;
}
__device__ __forceinline__ float gelu_exact(float x) {
  return 0.5f * x * (1.0f + erff(x * 0.70710678118654752f));
}

// 16-byte async DMA: global -> LDS, tracked by ASYNCcnt, no VGPR staging.
// LDS operand = low 32 bits of generic shared pointer (== LDS byte offset).
__device__ __forceinline__ void async_copy16(void* lds, const void* gmem) {
  unsigned loff = (unsigned)(size_t)lds;
  asm volatile("global_load_async_to_lds_b128 %0, %1, off"
               :: "v"(loff), "v"(gmem)
               : "memory");
}

// ---------------- LayerNorm (fp32 in -> bf16 out), one row per block ----------------
__global__ __launch_bounds__(256) void ln_bf16_kernel(const float* __restrict__ x,
                                                      const float* __restrict__ g,
                                                      const float* __restrict__ be,
                                                      __bf16* __restrict__ out) {
  __shared__ float red[8];
  const int row = blockIdx.x;
  const int c = threadIdx.x;
  const int lane = c & 31, wv = c >> 5;
  const float v = x[(size_t)row * C_ + c];
  float s = v;
#pragma unroll
  for (int m = 1; m < 32; m <<= 1) s += __shfl_xor(s, m, 32);
  if (lane == 0) red[wv] = s;
  __syncthreads();
  float tot = 0.f;
#pragma unroll
  for (int k = 0; k < 8; ++k) tot += red[k];
  const float mu = tot * (1.0f / C_);
  __syncthreads();
  const float d = v - mu;
  float s2 = d * d;
#pragma unroll
  for (int m = 1; m < 32; m <<= 1) s2 += __shfl_xor(s2, m, 32);
  if (lane == 0) red[wv] = s2;
  __syncthreads();
  float tv = 0.f;
#pragma unroll
  for (int k = 0; k < 8; ++k) tv += red[k];
  const float var = tv * (1.0f / C_);
  out[(size_t)row * C_ + c] = (__bf16)(d * rsqrtf(var + 1e-5f) * g[c] + be[c]);
}

// ---------------- Weight prep: fused QKV -> bf16 [3C][C] ----------------
__global__ void prep_wqkv_kernel(const float* __restrict__ Wq, const float* __restrict__ Wk,
                                 const float* __restrict__ Wv, __bf16* __restrict__ out) {
  const int idx = blockIdx.x * blockDim.x + threadIdx.x;
  if (idx >= 3 * C_ * C_) return;
  const int n = idx / C_, k = idx % C_;
  const int r = n >> 8, hh = (n >> 4) & 15, d = n & 15;
  const float* W = (r == 0) ? Wq : (r == 1) ? Wk : Wv;   // [H][C][HS]
  out[idx] = (__bf16)W[hh * (C_ * HS_) + k * HS_ + d];
}

// ---------------- Weight prep: transpose [K][N] fp32 -> [N][K] bf16 ----------------
__global__ void prep_tr_kernel(const float* __restrict__ W, __bf16* __restrict__ Wt,
                               int K, int N) {
  const int idx = blockIdx.x * blockDim.x + threadIdx.x;
  if (idx >= K * N) return;
  const int n = idx / K, k = idx % K;
  Wt[idx] = (__bf16)W[(size_t)k * N + n];
}

// ------- GEMM core: 128x64 block tile, 8 waves (16 rows x 64 cols each), ---------
// ------- async global->LDS double buffering, 4 WMMA per wave per K-step. ---------
#define ISSUE_TILE(Aptr, Bptr, Kdim, p, k0)                                             \
  async_copy16(&As[p][lr][lc],      &(Aptr)[(size_t)(bm + lr) * (Kdim) + (k0) + lc]);   \
  async_copy16(&As[p][64 + lr][lc], &(Aptr)[(size_t)(bm + 64 + lr) * (Kdim) + (k0) + lc]); \
  async_copy16(&Bs[p][lr][lc],      &(Bptr)[(size_t)(bn + lr) * (Kdim) + (k0) + lc]);

#define GEMM_CORE(Aptr, Bptr, Kdim)                                            \
  __shared__ __bf16 As[2][128][40];                                            \
  __shared__ __bf16 Bs[2][64][40];                                             \
  const int tid = threadIdx.x;                                                 \
  const int lane = tid & 31, wv = tid >> 5;                                    \
  const int hf = lane >> 4, ln16 = lane & 15;                                  \
  const int bm = blockIdx.x * 128, bn = blockIdx.y * 64;                       \
  const int lr = tid >> 2, lc = (tid & 3) * 8;                                 \
  v8f acc[4] = {{}, {}, {}, {}};                                               \
  int p = 0;                                                                   \
  ISSUE_TILE(Aptr, Bptr, Kdim, 0, 0)                                           \
  for (int k0 = 0; k0 < (Kdim); k0 += 32) {                                    \
    if (k0 + 32 < (Kdim)) {                                                    \
      ISSUE_TILE(Aptr, Bptr, Kdim, p ^ 1, k0 + 32)                             \
      asm volatile("s_wait_asynccnt 3" ::: "memory");                          \
    } else {                                                                   \
      asm volatile("s_wait_asynccnt 0" ::: "memory");                          \
    }                                                                          \
    __syncthreads();                                                           \
    AF fa, fb[4];                                                              \
    fa.q[0] = *(const uint4*)&As[p][wv * 16 + ln16][hf * 16];                  \
    fa.q[1] = *(const uint4*)&As[p][wv * 16 + ln16][hf * 16 + 8];              \
    _Pragma("unroll")                                                          \
    for (int t = 0; t < 4; ++t) {                                              \
      fb[t].q[0] = *(const uint4*)&Bs[p][t * 16 + ln16][hf * 16];              \
      fb[t].q[1] = *(const uint4*)&Bs[p][t * 16 + ln16][hf * 16 + 8];          \
    }                                                                          \
    _Pragma("unroll")                                                          \
    for (int t = 0; t < 4; ++t) acc[t] = WMMA_BF16(fa.v, fb[t].v, acc[t]);     \
    __syncthreads();                                                           \
    p ^= 1;                                                                    \
  }

// EPI 0: out_f32 = acc + bias[n] + resid[m][n];  EPI 1: out_bf16 = gelu(acc + bias[n])
template <int EPI>
__global__ __launch_bounds__(256) void gemm_bf16_kernel(
    const __bf16* __restrict__ A, const __bf16* __restrict__ Bt,
    const float* __restrict__ bias, const float* __restrict__ resid,
    float* __restrict__ outF, __bf16* __restrict__ outB, int N, int K) {
  GEMM_CORE(A, Bt, K)
#pragma unroll
  for (int j = 0; j < 8; ++j) {
    const int m = bm + wv * 16 + j + 8 * hf;
#pragma unroll
    for (int t = 0; t < 4; ++t) {
      const int n = bn + t * 16 + ln16;
      if (EPI == 0) {
        outF[(size_t)m * N + n] = acc[t][j] + bias[n] + resid[(size_t)m * N + n];
      } else {
        outB[(size_t)m * N + n] = (__bf16)gelu_exact(acc[t][j] + bias[n]);
      }
    }
  }
}

// Fused QKV GEMM: scatter epilogue -> q/k padded [B*H][T][32] bf16, vT [B*H][HS][T] bf16
__global__ __launch_bounds__(256) void gemm_qkv_kernel(
    const __bf16* __restrict__ A, const __bf16* __restrict__ Bt,
    __bf16* __restrict__ qp, __bf16* __restrict__ kp, __bf16* __restrict__ vT) {
  GEMM_CORE(A, Bt, C_)
#pragma unroll
  for (int j = 0; j < 8; ++j) {
    const int m = bm + wv * 16 + j + 8 * hf;
    const int bb = m >> 8, t8 = m & 255;
#pragma unroll
    for (int t = 0; t < 4; ++t) {
      const int n = bn + t * 16 + ln16;
      const float val = acc[t][j];
      const int r = n >> 8, hh = (n >> 4) & 15, d = n & 15;
      const size_t bh = (size_t)bb * H_ + hh;
      if (r == 0) {
        qp[(bh * T_ + t8) * 32 + d]      = (__bf16)val;
        qp[(bh * T_ + t8) * 32 + 16 + d] = (__bf16)0.0f;   // zero pad K=16..31
      } else if (r == 1) {
        kp[(bh * T_ + t8) * 32 + d]      = (__bf16)val;
        kp[(bh * T_ + t8) * 32 + 16 + d] = (__bf16)0.0f;
      } else {
        vT[(bh * HS_ + d) * T_ + t8] = (__bf16)val;
      }
    }
  }
}

// ---------------- Flash attention: 1 block per (b,h); 16 waves, 1 query tile each ----
__global__ __launch_bounds__(512) void attn_kernel(
    const __bf16* __restrict__ qp, const __bf16* __restrict__ kp,
    const __bf16* __restrict__ vT, __bf16* __restrict__ attnB) {
  __shared__ __bf16 Pst[16][16][40];   // per-wave P staging, padded rows
  const int tid = threadIdx.x;
  const int lane = tid & 31, w = tid >> 5;
  const int hf = lane >> 4, ln16 = lane & 15;
  const int bh = blockIdx.x;
  const int b = bh >> 4, h = bh & 15;
  const int i = w;                      // query tile index (0..15)
  const __bf16* qb = qp + (size_t)bh * T_ * 32;
  const __bf16* kb = kp + (size_t)bh * T_ * 32;
  const __bf16* vb = vT + (size_t)bh * HS_ * T_;

  AF qf;
  {
    const __bf16* qr = qb + (size_t)(i * 16 + ln16) * 32;
    qf.q[0] = *(const uint4*)(qr + hf * 16);
    qf.q[1] = *(const uint4*)(qr + hf * 16 + 8);
  }
  float mrun[8], lrun[8];
  v8f O = {};
#pragma unroll
  for (int j = 0; j < 8; ++j) { mrun[j] = -INFINITY; lrun[j] = 0.0f; }
  const float scale = 0.0625f;          // 1/sqrt(C), C = n_embd (reference quirk)

  for (int jj = 0; jj <= i; jj += 2) {  // 32 keys per step
    AF kf0, kf1;
    const __bf16* kr0 = kb + (size_t)(jj * 16 + ln16) * 32;
    const __bf16* kr1 = kb + (size_t)((jj + 1) * 16 + ln16) * 32;
    kf0.q[0] = *(const uint4*)(kr0 + hf * 16);
    kf0.q[1] = *(const uint4*)(kr0 + hf * 16 + 8);
    kf1.q[0] = *(const uint4*)(kr1 + hf * 16);
    kf1.q[1] = *(const uint4*)(kr1 + hf * 16 + 8);
    v8f z = {};
    v8f s0 = WMMA_BF16(qf.v, kf0.v, z);
    v8f s1 = WMMA_BF16(qf.v, kf1.v, z);
#pragma unroll
    for (int j = 0; j < 8; ++j) {
      const int qrow = i * 16 + j + 8 * hf;
      float a0 = s0[j] * scale;
      float a1 = s1[j] * scale;
      if (jj * 16 + ln16 > qrow)       a0 = -INFINITY;   // causal mask
      if ((jj + 1) * 16 + ln16 > qrow) a1 = -INFINITY;
      const float mx = half_max16(fmaxf(a0, a1));
      const float mn = fmaxf(mrun[j], mx);
      const float corr = expf(mrun[j] - mn);
      const float p0 = expf(a0 - mn);
      const float p1 = expf(a1 - mn);
      const float rs = half_sum16(p0 + p1);
      lrun[j] = lrun[j] * corr + rs;
      mrun[j] = mn;
      O[j] *= corr;
      Pst[w][j + 8 * hf][ln16]      = (__bf16)p0;
      Pst[w][j + 8 * hf][16 + ln16] = (__bf16)p1;
    }
    // cross-lane wave-internal LDS dependency: force DS completion + compiler barrier
    asm volatile("s_wait_dscnt 0" ::: "memory");
    AF pf, vf;
    const __bf16* pr = &Pst[w][ln16][0];
    pf.q[0] = *(const uint4*)(pr + hf * 16);
    pf.q[1] = *(const uint4*)(pr + hf * 16 + 8);
    const __bf16* vr = vb + (size_t)ln16 * T_ + jj * 16;
    vf.q[0] = *(const uint4*)(vr + hf * 16);
    vf.q[1] = *(const uint4*)(vr + hf * 16 + 8);
    O = WMMA_BF16(pf.v, vf.v, O);
  }
#pragma unroll
  for (int j = 0; j < 8; ++j) {
    const int row = i * 16 + j + 8 * hf;
    attnB[(size_t)(b * T_ + row) * C_ + h * 16 + ln16] = (__bf16)(O[j] / lrun[j]);
  }
}

// ---------------- Host launch ----------------
extern "C" void kernel_launch(void* const* d_in, const int* in_sizes, int n_in,
                              void* d_out, int out_size, void* d_ws, size_t ws_size,
                              hipStream_t stream) {
  (void)in_sizes; (void)n_in; (void)out_size; (void)ws_size;
  const float* x   = (const float*)d_in[0];
  const float* Wq  = (const float*)d_in[1];
  const float* Wk  = (const float*)d_in[2];
  const float* Wv  = (const float*)d_in[3];
  const float* Wp  = (const float*)d_in[4];
  const float* bp  = (const float*)d_in[5];
  const float* W1  = (const float*)d_in[6];
  const float* b1  = (const float*)d_in[7];
  const float* W2  = (const float*)d_in[8];
  const float* b2  = (const float*)d_in[9];
  const float* g1  = (const float*)d_in[10];
  const float* be1 = (const float*)d_in[11];
  const float* g2  = (const float*)d_in[12];
  const float* be2 = (const float*)d_in[13];
  float* out = (float*)d_out;

  char* ws = (char*)d_ws;
  size_t o = 0;
  auto alloc = [&](size_t bytes) {
    char* p = ws + o;
    o += (bytes + 255) & ~(size_t)255;
    return p;
  };
  __bf16* h1    = (__bf16*)alloc((size_t)M_ * C_ * 2);
  __bf16* qp    = (__bf16*)alloc((size_t)B_ * H_ * T_ * 32 * 2);
  __bf16* kp    = (__bf16*)alloc((size_t)B_ * H_ * T_ * 32 * 2);
  __bf16* vTt   = (__bf16*)alloc((size_t)B_ * H_ * HS_ * T_ * 2);
  __bf16* attnB = (__bf16*)alloc((size_t)M_ * C_ * 2);
  float*  x2    = (float*)alloc((size_t)M_ * C_ * 4);
  __bf16* h2    = (__bf16*)alloc((size_t)M_ * C_ * 2);
  __bf16* ffn1  = (__bf16*)alloc((size_t)M_ * FF_ * 2);
  __bf16* WqkvT = (__bf16*)alloc((size_t)3 * C_ * C_ * 2);
  __bf16* WpT   = (__bf16*)alloc((size_t)C_ * C_ * 2);
  __bf16* W1T   = (__bf16*)alloc((size_t)C_ * FF_ * 2);
  __bf16* W2T   = (__bf16*)alloc((size_t)FF_ * C_ * 2);

  // weight conversion (deterministic, every call)
  prep_wqkv_kernel<<<(3 * C_ * C_ + 255) / 256, 256, 0, stream>>>(Wq, Wk, Wv, WqkvT);
  prep_tr_kernel<<<(C_ * C_ + 255) / 256, 256, 0, stream>>>(Wp, WpT, C_, C_);
  prep_tr_kernel<<<(C_ * FF_ + 255) / 256, 256, 0, stream>>>(W1, W1T, C_, FF_);
  prep_tr_kernel<<<(FF_ * C_ + 255) / 256, 256, 0, stream>>>(W2, W2T, FF_, C_);

  // 1) h1 = LN(x) in bf16
  ln_bf16_kernel<<<M_, 256, 0, stream>>>(x, g1, be1, h1);
  // 2) fused QKV projection
  gemm_qkv_kernel<<<dim3(M_ / 128, (3 * C_) / 64), 256, 0, stream>>>(h1, WqkvT, qp, kp, vTt);
  // 3) causal flash attention -> concatenated heads (bf16)
  attn_kernel<<<B_ * H_, 512, 0, stream>>>(qp, kp, vTt, attnB);
  // 4) x2 = x + attn @ Wp + bp
  gemm_bf16_kernel<0><<<dim3(M_ / 128, C_ / 64), 256, 0, stream>>>(
      attnB, WpT, bp, x, x2, (__bf16*)nullptr, C_, C_);
  // 5) h2 = LN(x2)
  ln_bf16_kernel<<<M_, 256, 0, stream>>>(x2, g2, be2, h2);
  // 6) ffn1 = gelu(h2 @ W1 + b1) in bf16
  gemm_bf16_kernel<1><<<dim3(M_ / 128, FF_ / 64), 256, 0, stream>>>(
      h2, W1T, b1, (const float*)nullptr, (float*)nullptr, ffn1, FF_, C_);
  // 7) out = x2 + ffn1 @ W2 + b2
  gemm_bf16_kernel<0><<<dim3(M_ / 128, C_ / 64), 256, 0, stream>>>(
      ffn1, W2T, b2, x2, out, (__bf16*)nullptr, C_, FF_);
}
